// Model_DH_84464826843605
// MI455X (gfx1250) — matrix operand order repaired and checked by reading, not compiled, not run on hardware
//
#include <hip/hip_runtime.h>
#include <hip/hip_bf16.h>

// Problem constants (from reference)
#define B_    128
#define T_    1000
#define IN_   700
#define H_    512
#define OUT_  20
#define KP_   704            // IN padded to multiple of 32 for 16x16x32 bf16 WMMA
#define M_    (B_ * T_)      // 128000 GEMM rows
#define NCHUNK_ (KP_ / 32)   // 22 K-chunks

typedef __bf16 v16bf __attribute__((ext_vector_type(16)));
typedef __bf16 v8bf  __attribute__((ext_vector_type(8)));
typedef float  v8f   __attribute__((ext_vector_type(8)));
typedef float  v4f   __attribute__((ext_vector_type(4)));
typedef int    v4i   __attribute__((ext_vector_type(4)));

// ---- gfx1250 async-copy (memory -> LDS, tracked by ASYNCcnt) --------------
#if defined(__has_builtin)
#  if __has_builtin(__builtin_amdgcn_global_load_async_to_lds_b128)
#    define HAVE_ASYNC_COPY 1
#  endif
#  if __has_builtin(__builtin_amdgcn_s_wait_asynccnt)
#    define HAVE_WAIT_ASYNC 1
#  endif
#endif

__device__ __forceinline__ void async_copy16(const void* g, void* l) {
#if defined(HAVE_ASYNC_COPY)
  // Per-lane 16B copy, global -> LDS, no VGPR data; lands on ASYNCcnt.
  // Builtin signature (from clang diagnostic): (as1 int4*, as3 int4*, Ii, Ii)
  __builtin_amdgcn_global_load_async_to_lds_b128(
      (__attribute__((address_space(1))) v4i*)g,
      (__attribute__((address_space(3))) v4i*)l, 0, 0);
#else
  *(v4i*)l = *(const v4i*)g;  // sync fallback (LDS store, DScnt)
#endif
}

__device__ __forceinline__ void wait_async_copies() {
#if defined(HAVE_ASYNC_COPY)
#  if defined(HAVE_WAIT_ASYNC)
  __builtin_amdgcn_s_wait_asynccnt(0);
#  else
  asm volatile("s_wait_asynccnt 0x0" ::: "memory");
#  endif
#endif
}

__device__ __forceinline__ float sigmoidf_(float x) {
  return 1.0f / (1.0f + __expf(-x));
}

// ---------------------------------------------------------------------------
// Kernel 0: convert x [M_,IN_] and W1 [H_,IN_] (f32) to bf16, K zero-padded
// to KP_. One block per row; 4 elements per thread.
// ---------------------------------------------------------------------------
__global__ void prep_bf16(const float* __restrict__ x,
                          const float* __restrict__ W1,
                          __bf16* __restrict__ xbf,
                          __bf16* __restrict__ w1bf) {
  const int row = blockIdx.x;
  const int k = threadIdx.x * 4;
  if (k >= KP_) return;
  const float* src;
  __bf16* dst;
  if (row < M_) {
    src = x + (size_t)row * IN_;
    dst = xbf + (size_t)row * KP_;
  } else {
    const int r = row - M_;
    src = W1 + (size_t)r * IN_;
    dst = w1bf + (size_t)r * KP_;
  }
#pragma unroll
  for (int j = 0; j < 4; ++j) {
    const int kk = k + j;
    const float v = (kk < IN_) ? src[kk] : 0.0f;
    dst[kk] = (__bf16)v;
  }
}

// ---------------------------------------------------------------------------
// Kernel 1: proj[m][n] = sum_k xbf[m][k] * w1bf[n][k] + b1[n]   (f32 out)
//
// 256 threads = 8 waves; each wave computes a 16(M) x 128(N) strip as 8
// chained V_WMMA_F32_16X16X32_BF16 tiles. The full 512x32 bf16 B-chunk of W1
// is staged into a double-buffered LDS tile with async global->LDS copies
// (ASYNCcnt), issued one chunk ahead so the copy overlaps the WMMA chain.
// Waves w and w+4 (same n-strips, different m-strips) share the LDS tile,
// halving global B traffic vs direct loads.
//
// A fragment (16-bit A 16x32): lane l (half = l>>4) of row m=l&15
// holds K = {c*32 + half*16 .. +15} -> one contiguous 32B global load.
// B fragment (32x16): lane l holds column n=l&15, K = {c*32+half*16 .. +15}
// -> one contiguous 32B LDS read (row stride 64B keeps it 32B-aligned).
// ---------------------------------------------------------------------------
static __device__ __forceinline__ void stage_B(const __bf16* __restrict__ w1bf,
                                               __bf16 (*bt)[32], int c,
                                               int tid) {
  // 512 rows x 32 bf16 = 2048 x 16B transfers; 8 per thread.
#pragma unroll
  for (int j = 0; j < 8; ++j) {
    const int idx = tid + j * 256;  // 0..2047
    const int n = idx >> 2;
    const int sub = idx & 3;
    async_copy16(w1bf + (size_t)n * KP_ + c * 32 + sub * 8, &bt[n][sub * 8]);
  }
}

__global__ __launch_bounds__(256) void gemm_proj(
    const __bf16* __restrict__ xbf, const __bf16* __restrict__ w1bf,
    const float* __restrict__ b1, float* __restrict__ proj) {
  __shared__ __bf16 Bt[2][H_][32];  // 64 KB double-buffered B chunk

  const int tid = threadIdx.x;
  const int wave = tid >> 5;
  const int lane = tid & 31;
  const int lmod = lane & 15;
  const int half = lane >> 4;

  const int mstrip = blockIdx.x * 2 + (wave >> 2);  // 2 M-strips per block
  const int nstrip = wave & 3;                      // 4 N-strips of 128
  const int m0 = mstrip * 16;
  const int n0 = nstrip * 128;

  v8f acc[8];
#pragma unroll
  for (int i = 0; i < 8; ++i)
    acc[i] = (v8f){0.f, 0.f, 0.f, 0.f, 0.f, 0.f, 0.f, 0.f};

  const __bf16* __restrict__ arow = xbf + (size_t)(m0 + lmod) * KP_;

  // Prologue: stage chunk 0
  stage_B(w1bf, Bt[0], 0, tid);
  wait_async_copies();
  __syncthreads();

#pragma unroll 1
  for (int c = 0; c < NCHUNK_; ++c) {
    const int cur = c & 1;
    if (c + 1 < NCHUNK_) stage_B(w1bf, Bt[cur ^ 1], c + 1, tid);  // prefetch

    const int kb = c * 32;
    const v16bf afrag = *(const v16bf*)(arow + kb + half * 16);

#pragma unroll
    for (int nt = 0; nt < 8; ++nt) {
      const v16bf bfrag =
          *(const v16bf*)&Bt[cur][n0 + nt * 16 + lmod][half * 16];
      acc[nt] = __builtin_amdgcn_wmma_f32_16x16x32_bf16(
          false, afrag, false, bfrag, (short)0, acc[nt], false, false);
    }

    wait_async_copies();  // staging for c+1 complete (this wave's ASYNCcnt)
    __syncthreads();      // all waves done reading Bt[cur] / staging Bt[cur^1]
  }

  // Epilogue: D layout (lane n=l&15; VGPR r -> row r + 8*half) + bias, f32 out
#pragma unroll
  for (int nt = 0; nt < 8; ++nt) {
    const int n = n0 + nt * 16 + lmod;
    const float bias = b1[n];
#pragma unroll
    for (int r = 0; r < 8; ++r) {
      proj[(size_t)(m0 + r + 8 * half) * H_ + n] = acc[nt][r] + bias;
    }
  }
}

// ---------------------------------------------------------------------------
// Kernel 2: sequential DH-SNN scan. One block (256 thr / 8 waves) per 16
// batch rows. Wave w updates neuron state for rows {2w, 2w+1}; each lane owns
// 16 hidden units per row, fully register-resident for all T steps. Spikes go
// to LDS as bf16 (exact); waves 0/1 run the 16x512x32 readout with bf16 WMMA
// (W2 LDS-resident as bf16, OUT padded 20->32), mem2 lives in WMMA D layout.
// ---------------------------------------------------------------------------
__global__ __launch_bounds__(256) void dhsnn_scan(
    const float* __restrict__ proj,
    const float* __restrict__ tau_m1, const float* __restrict__ tau_n1,
    const float* __restrict__ W2, const float* __restrict__ b2,
    const float* __restrict__ tau_m2,
    const float* __restrict__ mem1_0, const float* __restrict__ mem2_0,
    float* __restrict__ out) {
  __shared__ __bf16 spkLDS[16][H_];   // 16 KB: spike matrix (batch x hidden)
  __shared__ __bf16 w2LDS[32][H_];    // 32 KB: W2 (zero-padded to 32 rows)

  const int tid = threadIdx.x;
  const int wave = tid >> 5;
  const int lane = tid & 31;
  const int lmod = lane & 15;
  const int half = lane >> 4;
  const int bbase = blockIdx.x * 16;

  // Stage W2 -> LDS as bf16 (rows 20..31 zero)
  for (int i = tid; i < 32 * H_; i += 256) {
    const int r = i >> 9;
    const int cc = i & (H_ - 1);
    const float v = (r < OUT_) ? W2[r * H_ + cc] : 0.0f;
    w2LDS[r][cc] = (__bf16)v;
  }

  // Per-lane decay constants for its 16 hidden units
  const int hb = lane * 16;
  float a1[16], be1[16];
#pragma unroll
  for (int j = 0; j < 16; ++j) {
    a1[j] = sigmoidf_(tau_m1[hb + j]);
    be1[j] = sigmoidf_(tau_n1[hb + j]);
  }

  // Register-resident neuron state: 2 batch rows x 16 hidden units
  float din[2][16], mem1[2][16], spk[2][16];
#pragma unroll
  for (int rr = 0; rr < 2; ++rr) {
    const int b = bbase + wave * 2 + rr;
#pragma unroll
    for (int j = 0; j < 16; ++j) {
      din[rr][j] = 0.0f;
      spk[rr][j] = 0.0f;
      mem1[rr][j] = mem1_0[(size_t)b * H_ + hb + j];
    }
  }

  // Readout state (waves 0/1 only): n-tile = wave, mem2 in WMMA D layout
  const int nglob = wave * 16 + lmod;
  const bool valid = (wave < 2) && (nglob < OUT_);
  float a2 = 0.0f, b2v = 0.0f;
  v8f m2 = {0.f, 0.f, 0.f, 0.f, 0.f, 0.f, 0.f, 0.f};
  if (valid) {
    a2 = sigmoidf_(tau_m2[nglob]);
    b2v = b2[nglob];
#pragma unroll
    for (int r = 0; r < 8; ++r)
      m2[r] = mem2_0[(size_t)(bbase + r + 8 * half) * OUT_ + nglob];
  }

  __syncthreads();

#pragma unroll 1
  for (int t = 0; t < T_; ++t) {
    // --- LIF / dendrite state update (all 8 waves) ---
#pragma unroll
    for (int rr = 0; rr < 2; ++rr) {
      const int row = wave * 2 + rr;
      const int b = bbase + row;
      const float* p = proj + ((size_t)b * T_ + t) * H_ + hb;
      if (t + 1 < T_) __builtin_prefetch(p + H_, 0, 0);  // next timestep slice
      v4f pv[4];
#pragma unroll
      for (int q = 0; q < 4; ++q) pv[q] = *(const v4f*)(p + q * 4);
#pragma unroll
      for (int j = 0; j < 16; ++j) {
        const float pt = pv[j >> 2][j & 3];
        const float d = be1[j] * din[rr][j] + (1.0f - be1[j]) * pt;
        const float m =
            mem1[rr][j] * a1[j] + (1.0f - a1[j]) * d - spk[rr][j];  // VTH=1
        const float s = ((m - 1.0f) > 0.0f) ? 1.0f : 0.0f;
        din[rr][j] = d;
        mem1[rr][j] = m;
        spk[rr][j] = s;
        spkLDS[row][hb + j] = (__bf16)s;  // exact in bf16
      }
    }
    __syncthreads();

    // --- readout: y = spike @ W2^T via bf16 WMMA (waves 0,1; EXEC all-1s) ---
    if (wave < 2) {
      v8f y = {0.f, 0.f, 0.f, 0.f, 0.f, 0.f, 0.f, 0.f};
      const __bf16* __restrict__ abase = &spkLDS[lmod][0];
      const __bf16* __restrict__ bbp = &w2LDS[wave * 16 + lmod][0];
#pragma unroll 4
      for (int c = 0; c < H_ / 32; ++c) {
        const int kb = c * 32;
        const v16bf afrag = *(const v16bf*)(abase + kb + half * 16);
        const v16bf bfrag = *(const v16bf*)(bbp + kb + half * 16);
        y = __builtin_amdgcn_wmma_f32_16x16x32_bf16(
            false, afrag, false, bfrag, (short)0, y, false, false);
      }
      if (valid) {
#pragma unroll
        for (int r = 0; r < 8; ++r) {
          const float nm2 = a2 * m2[r] + (1.0f - a2) * (y[r] + b2v);
          m2[r] = nm2;
          out[((size_t)(bbase + r + 8 * half) * T_ + t) * OUT_ + nglob] = nm2;
        }
      }
    }
    __syncthreads();  // spkLDS reused next step
  }
}

// ---------------------------------------------------------------------------
// Launch. Workspace layout (all offsets 256B aligned):
//   [0)                xbf  : M_*KP_ bf16   = 180,224,000 B
//   [180,224,000)      w1bf : H_*KP_ bf16   =     720,896 B
//   [180,944,896)      proj : M_*H_ f32     = 262,144,000 B
//   total ~ 443 MB
// ---------------------------------------------------------------------------
extern "C" void kernel_launch(void* const* d_in, const int* in_sizes, int n_in,
                              void* d_out, int out_size, void* d_ws,
                              size_t ws_size, hipStream_t stream) {
  const float* x      = (const float*)d_in[0];
  const float* W1     = (const float*)d_in[1];
  const float* b1     = (const float*)d_in[2];
  const float* tau_m1 = (const float*)d_in[3];
  const float* tau_n1 = (const float*)d_in[4];
  const float* W2     = (const float*)d_in[5];
  const float* b2     = (const float*)d_in[6];
  const float* tau_m2 = (const float*)d_in[7];
  const float* mem1_0 = (const float*)d_in[8];
  const float* mem2_0 = (const float*)d_in[9];
  float* out = (float*)d_out;

  char* ws = (char*)d_ws;
  __bf16* xbf  = (__bf16*)(ws);
  __bf16* w1bf = (__bf16*)(ws + 180224000);
  float*  proj = (float*)(ws + 180944896);

  // 1) f32 -> bf16 staging (x rows then W1 rows), K zero-padded to 704
  prep_bf16<<<M_ + H_, 192, 0, stream>>>(x, W1, xbf, w1bf);

  // 2) proj = x @ W1^T + b1 : 128000 x 512 x 704 bf16 WMMA GEMM,
  //    async-to-LDS double-buffered B staging
  gemm_proj<<<M_ / 32, 256, 0, stream>>>(xbf, w1bf, b1, proj);

  // 3) sequential scan with register-resident state + WMMA readout
  dhsnn_scan<<<B_ / 16, 256, 0, stream>>>(proj, tau_m1, tau_n1, W2, b2,
                                          tau_m2, mem1_0, mem2_0, out);
}
